// BasicLSTMActorCritic_84756884620084
// MI455X (gfx1250) — matrix operand order repaired and checked by reading, not compiled, not run on hardware
//
#include <hip/hip_runtime.h>
#include <hip/hip_bf16.h>

// ---------------------------------------------------------------------------
// CDNA5 (gfx1250) LSTM actor-critic forward.
// Dims: T=512, B=256, O=720, H=128, A=2.
//   k_cvt   : fp32 -> f16 weight conversion (Wx, Wh, Wa1, Wc1)
//   k_gemm  : xW[T*B,512] = x[T*B,720] @ Wx_f16 + b   (WMMA f32_16x16x32_f16)
//             K loop peeled: 22 unconditional tiles + 1 guarded tail tile.
//   k_lstm  : 16 persistent workgroups x 16 batch rows; Wh resident in LDS in
//             fragment-swizzled order (2x ds_load_b128 per WMMA operand).
//   k_mlp1  : z = tanh(feat @ W1 + b1) for actor & critic branches (WMMA)
//   k_head2 : actor_mean (N=2), value (N=1), std=exp(log_std) scalar pass.
// ---------------------------------------------------------------------------

typedef __attribute__((ext_vector_type(16))) _Float16 v16h;
typedef __attribute__((ext_vector_type(8)))  _Float16 v8h;
typedef __attribute__((ext_vector_type(8)))  float    v8f;
typedef _Float16 half_t;

__device__ __forceinline__ v8f wmma_f16(v16h a, v16h b, v8f c) {
  // (neg_a, A, neg_b, B, c_mod, C, reuse_a, reuse_b)
  return __builtin_amdgcn_wmma_f32_16x16x32_f16(false, a, false, b, (short)0, c,
                                                false, false);
}

// A-matrix 16x32 f16 fragment: per lane, 16 contiguous halves:
// row[kb..kb+15], kb = 16*(lane>=16).
// Requires base 16B-aligned and ld a multiple of 8 halves.
__device__ __forceinline__ v16h load_a_frag(const half_t* base, int ld, int lane) {
  const int m  = lane & 15;
  const int kb = (lane & 16) ? 16 : 0;
  const half_t* row = base + m * ld + kb;
  const v8h lo = *(const v8h*)(row);
  const v8h hi = *(const v8h*)(row + 8);
  return __builtin_shufflevector(lo, hi, 0, 1, 2, 3, 4, 5, 6, 7,
                                 8, 9, 10, 11, 12, 13, 14, 15);
}

// B fragment from fragment-swizzled storage: 16 contiguous halves per lane.
__device__ __forceinline__ v16h load_frag_contig(const half_t* p) {
  const v8h lo = *(const v8h*)(p);
  const v8h hi = *(const v8h*)(p + 8);
  return __builtin_shufflevector(lo, hi, 0, 1, 2, 3, 4, 5, 6, 7,
                                 8, 9, 10, 11, 12, 13, 14, 15);
}

__device__ __forceinline__ float sigm(float x) { return 1.0f / (1.0f + __expf(-x)); }

// ---------------------------------------------------------------------------
// Weight fp32 -> f16 conversion
// ---------------------------------------------------------------------------
__global__ void __launch_bounds__(256)
k_cvt(const float* __restrict__ Wx, const float* __restrict__ Wh,
      const float* __restrict__ Wa1, const float* __restrict__ Wc1,
      half_t* __restrict__ wx_h, half_t* __restrict__ wh_h,
      half_t* __restrict__ wa1_h, half_t* __restrict__ wc1_h) {
  const int i = blockIdx.x * 256 + threadIdx.x;
  if (i < 368640)              wx_h[i]            = (half_t)Wx[i];
  else if (i < 434176)         wh_h[i - 368640]   = (half_t)Wh[i - 368640];
  else if (i < 450560)         wa1_h[i - 434176]  = (half_t)Wa1[i - 434176];
  else if (i < 466944)         wc1_h[i - 450560]  = (half_t)Wc1[i - 450560];
}

// ---------------------------------------------------------------------------
// Tile staging for k_gemm. FULL=true: no bounds checks (k0+32 <= 720).
// ---------------------------------------------------------------------------
template <bool FULL>
__device__ __forceinline__ void gemm_stage(const float* __restrict__ x,
                                           const half_t* __restrict__ wx,
                                           half_t* As, half_t* Bsw,
                                           size_t row_base, int bn, int k0,
                                           int t) {
  // Stage A (fp32 -> f16): 128x32, float4 per thread x4 rows.
  {
    const int q = t & 7, r0 = t >> 3;
    const bool kvalid = FULL || (k0 + q * 4) < 720;
#pragma unroll
    for (int i = 0; i < 4; ++i) {
      const int r = r0 + i * 32;
      float4 v = make_float4(0.f, 0.f, 0.f, 0.f);
      if (kvalid)
        v = *(const float4*)(x + (row_base + r) * 720 + k0 + q * 4);
      half_t* dst = As + r * 40 + q * 4;
      dst[0] = (half_t)v.x; dst[1] = (half_t)v.y;
      dst[2] = (half_t)v.z; dst[3] = (half_t)v.w;
    }
  }
  // Stage B swizzled: each thread handles 2 rows x 8 cols of the 32x128 tile.
  {
    const int c8 = t & 15, r0 = t >> 4;
    const int nt = c8 >> 1;                 // n-tile 0..7
    const int n0 = (c8 & 1) * 8;            // col within tile: 0 or 8
#pragma unroll
    for (int i = 0; i < 2; ++i) {
      const int r = r0 + i * 16;            // k within tile 0..31
      v8h v = {};
      if (FULL || (k0 + r) < 720)
        v = *(const v8h*)(wx + (size_t)(k0 + r) * 512 + bn * 128 + c8 * 8);
      const int lane_base = n0 + ((r & 16) ? 16 : 0);
      const int h = r & 15;
#pragma unroll
      for (int j = 0; j < 8; ++j)
        Bsw[((nt * 32 + lane_base + j) << 4) + h] = v[j];
    }
  }
}

// ---------------------------------------------------------------------------
// xW = x @ Wx + b.  M=131072, K=720, N=512.  128x128 tile / workgroup,
// 8 waves, each wave: 1 M-tile x 8 N-tiles of 16x16, Kc=32 per WMMA.
// B tile staged into LDS pre-swizzled into fragment order:
//   Bsw[((nt*32 + lane) << 4) + h], lane = n + 16*(k>=16), h = k&15.
// ---------------------------------------------------------------------------
__global__ void __launch_bounds__(256)
k_gemm(const float* __restrict__ x, const half_t* __restrict__ wx,
       const float* __restrict__ bias, float* __restrict__ xW) {
  __shared__ __align__(16) half_t As[128 * 40];     // 128 rows x 32 K (+8 pad)
  __shared__ __align__(16) half_t Bsw[8 * 32 * 16]; // 8 frags x 32 lanes x 16 h

  const int bn = blockIdx.x;           // 0..3
  const int bm = blockIdx.y;           // 0..1023
  const int t  = threadIdx.x;
  const int lane = t & 31, wave = t >> 5;
  const size_t row_base = (size_t)bm * 128;

  v8f vzero = {};
  v8f acc[8];
#pragma unroll
  for (int i = 0; i < 8; ++i) acc[i] = vzero;

  // 22 full K tiles (k0 = 0..672), then one guarded tail tile (k0 = 704).
  for (int k0 = 0; k0 < 704; k0 += 32) {
    gemm_stage<true>(x, wx, As, Bsw, row_base, bn, k0, t);
    __syncthreads();
    const v16h a = load_a_frag(As + wave * 16 * 40, 40, lane);
#pragma unroll
    for (int nt = 0; nt < 8; ++nt) {
      const v16h bf = load_frag_contig(Bsw + ((nt * 32 + lane) << 4));
      acc[nt] = wmma_f16(a, bf, acc[nt]);
    }
    __syncthreads();
  }
  {
    gemm_stage<false>(x, wx, As, Bsw, row_base, bn, 704, t);
    __syncthreads();
    const v16h a = load_a_frag(As + wave * 16 * 40, 40, lane);
#pragma unroll
    for (int nt = 0; nt < 8; ++nt) {
      const v16h bf = load_frag_contig(Bsw + ((nt * 32 + lane) << 4));
      acc[nt] = wmma_f16(a, bf, acc[nt]);
    }
    __syncthreads();
  }

  // Store + bias. C layout: n = lane&15, m = v + 8*(lane>=16).
  const int nlane = lane & 15, mb = (lane & 16) ? 8 : 0;
#pragma unroll
  for (int nt = 0; nt < 8; ++nt) {
    const int col = bn * 128 + nt * 16 + nlane;
    const float bv = bias[col];
#pragma unroll
    for (int v = 0; v < 8; ++v) {
      const size_t row = row_base + (size_t)wave * 16 + mb + v;
      xW[row * 512 + col] = acc[nt][v] + bv;
    }
  }
}

// ---------------------------------------------------------------------------
// LSTM recurrence. 16 workgroups x 16 batch rows. Wh (fragment-swizzled),
// h (f16+f32) and c resident in LDS for all 512 steps. Per step, wave w owns
// hidden cols [w*16, w*16+16): i/f/g/o tiles, K=128 -> 16 WMMAs/wave/step.
// Whsw frag id = ntile*4 + kt, ntile = g*8 + w.
// ---------------------------------------------------------------------------
__global__ void __launch_bounds__(256)
k_lstm(const float* __restrict__ xW, const half_t* __restrict__ wh,
       const float* __restrict__ h0, const float* __restrict__ c0,
       float* __restrict__ out_hT, float* __restrict__ out_cT,
       half_t* __restrict__ feat_h) {
  extern __shared__ char smem[];
  half_t* Whsw = (half_t*)smem;                                  // 128*512 h
  half_t* hh   = (half_t*)(smem + 128 * 512 * 2);                // 16 x 136
  float*  hf   = (float*)(smem + 128 * 512 * 2 + 16 * 136 * 2);  // 16 x 128
  float*  cf   = hf + 16 * 128;                                  // 16 x 128

  const int t = threadIdx.x;
  const int lane = t & 31, wave = t >> 5;
  const int b0 = blockIdx.x * 16;

  // Stage Wh (128x512 f16) into fragment-swizzled LDS.
#pragma unroll 2
  for (int i = 0; i < 32; ++i) {
    const int idx = t + i * 256;             // 8192 chunks of 8 halves
    const int r = idx >> 6, c8 = idx & 63;   // r: k row 0..127
    const v8h v = *(const v8h*)(wh + (size_t)r * 512 + c8 * 8);
    const int nt = c8 >> 1;                  // n-tile 0..31
    const int n0 = (c8 & 1) * 8;
    const int kt = r >> 5;
    const int lane_base = n0 + ((r & 16) ? 16 : 0);
    const int h = r & 15;
    const int frag = nt * 4 + kt;
#pragma unroll
    for (int j = 0; j < 8; ++j)
      Whsw[((frag * 32 + lane_base + j) << 4) + h] = v[j];
  }
  // Init h, c state.
  for (int idx = t; idx < 16 * 128; idx += 256) {
    const int r = idx >> 7, cc = idx & 127;
    const float hv = h0[(b0 + r) * 128 + cc];
    const float cv = c0[(b0 + r) * 128 + cc];
    hf[idx] = hv;
    cf[idx] = cv;
    hh[r * 136 + cc] = (half_t)hv;
  }
  __syncthreads();

  const int nlane = lane & 15, mb = (lane & 16) ? 8 : 0;
  const int hc = wave * 16 + nlane;

  for (int step = 0; step < 512; ++step) {
    const float* xw_base = xW + ((size_t)step * 256 + b0) * 512;

    // Init accumulators from xW (gate pre-activations), gate g at col g*128+w*16.
    v8f acc[4];
#pragma unroll
    for (int g = 0; g < 4; ++g) {
      const float* p = xw_base + (size_t)mb * 512 + g * 128 + wave * 16 + nlane;
#pragma unroll
      for (int v = 0; v < 8; ++v) acc[g][v] = p[(size_t)v * 512];
    }
    // Prefetch next timestep's xW slice toward L2 (global_prefetch_b8).
    if (step + 1 < 512)
      __builtin_prefetch(xw_base + (size_t)256 * 512 + t * 32, 0, 1);

    // Load all A fragments of h (K = 0..127) before anyone overwrites h.
    v16h af[4];
#pragma unroll
    for (int kt = 0; kt < 4; ++kt)
      af[kt] = load_a_frag(hh + kt * 32, 136, lane);
    __syncthreads();   // all h reads complete

#pragma unroll
    for (int g = 0; g < 4; ++g) {
      const int ntb = (g * 8 + wave) * 4;
#pragma unroll
      for (int kt = 0; kt < 4; ++kt) {
        const v16h bf = load_frag_contig(Whsw + (((ntb + kt) * 32 + lane) << 4));
        acc[g] = wmma_f16(af[kt], bf, acc[g]);
      }
    }

    // Elementwise cell update; each lane owns unique (m, hc).
#pragma unroll
    for (int v = 0; v < 8; ++v) {
      const int m = mb + v;
      const float iv = sigm(acc[0][v]);
      const float fv = sigm(acc[1][v]);
      const float gv = tanhf(acc[2][v]);
      const float ov = sigm(acc[3][v]);
      const float cn = fv * cf[m * 128 + hc] + iv * gv;
      const float hn = ov * tanhf(cn);
      cf[m * 128 + hc] = cn;
      hf[m * 128 + hc] = hn;
      hh[m * 136 + hc] = (half_t)hn;
    }
    __syncthreads();   // h writes visible before next step's reads
  }

  // Emit hT, cT (fp32 outputs) and feat (f16 for the head GEMMs).
  for (int idx = t; idx < 16 * 128; idx += 256) {
    const int r = idx >> 7, cc = idx & 127;
    out_hT[(b0 + r) * 128 + cc] = hf[idx];
    out_cT[(b0 + r) * 128 + cc] = cf[idx];
    feat_h[(b0 + r) * 128 + cc] = hh[r * 136 + cc];
  }
}

// ---------------------------------------------------------------------------
// z = tanh(feat @ W1 + b1).  M=256, K=128, N=128.  block 0: actor, 1: critic.
// Weight staged fragment-swizzled: frag id = nt*4 + kt (nt 0..7, kt 0..3).
// ---------------------------------------------------------------------------
__global__ void __launch_bounds__(256)
k_mlp1(const half_t* __restrict__ feat_h,
       const half_t* __restrict__ wa1_h, const half_t* __restrict__ wc1_h,
       const float* __restrict__ ba1, const float* __restrict__ bc1,
       float* __restrict__ z1, float* __restrict__ zc1) {
  __shared__ __align__(16) half_t Wsw[32 * 32 * 16];  // 32 frags x 32 x 16
  const half_t* W   = (blockIdx.x == 0) ? wa1_h : wc1_h;
  const float* bias = (blockIdx.x == 0) ? ba1 : bc1;
  float* zout       = (blockIdx.x == 0) ? z1 : zc1;

  const int t = threadIdx.x;
  for (int i = t; i < 128 * 16; i += 256) {   // 2048 chunks of 8 halves
    const int r = i >> 4, c8 = i & 15;        // r: k row 0..127
    const v8h v = *(const v8h*)(W + (size_t)r * 128 + c8 * 8);
    const int nt = c8 >> 1;
    const int n0 = (c8 & 1) * 8;
    const int kt = r >> 5;
    const int lane_base = n0 + ((r & 16) ? 16 : 0);
    const int h = r & 15;
    const int frag = nt * 4 + kt;
#pragma unroll
    for (int j = 0; j < 8; ++j)
      Wsw[((frag * 32 + lane_base + j) << 4) + h] = v[j];
  }
  __syncthreads();

  const int lane = t & 31, wave = t >> 5;
  const int nlane = lane & 15, mb = (lane & 16) ? 8 : 0;

#pragma unroll
  for (int mi = 0; mi < 2; ++mi) {
    const int mt = wave * 2 + mi;
    v16h af[4];
#pragma unroll
    for (int kt = 0; kt < 4; ++kt)
      af[kt] = load_a_frag(feat_h + (size_t)mt * 16 * 128 + kt * 32, 128, lane);
    for (int nt = 0; nt < 8; ++nt) {
      v8f acc = {};
#pragma unroll
      for (int kt = 0; kt < 4; ++kt) {
        const v16h bf =
            load_frag_contig(Wsw + (((nt * 4 + kt) * 32 + lane) << 4));
        acc = wmma_f16(af[kt], bf, acc);
      }
      const float bv = bias[nt * 16 + nlane];
#pragma unroll
      for (int v = 0; v < 8; ++v)
        zout[(size_t)(mt * 16 + mb + v) * 128 + nt * 16 + nlane] =
            tanhf(acc[v] + bv);
    }
  }
}

// ---------------------------------------------------------------------------
// Final projections: actor_mean[256,2], value[256], std[2].
// ---------------------------------------------------------------------------
__global__ void __launch_bounds__(256)
k_head2(const float* __restrict__ z1, const float* __restrict__ zc1,
        const float* __restrict__ Wa2, const float* __restrict__ ba2,
        const float* __restrict__ log_std, const float* __restrict__ Wc2,
        const float* __restrict__ bc2, float* __restrict__ out) {
  const int r = threadIdx.x;   // 0..255 = batch row
  float s0 = ba2[0], s1 = ba2[1], sv = bc2[0];
#pragma unroll 4
  for (int k = 0; k < 128; ++k) {
    const float za = z1[r * 128 + k];
    s0 += za * Wa2[k * 2 + 0];
    s1 += za * Wa2[k * 2 + 1];
    sv += zc1[r * 128 + k] * Wc2[k];
  }
  out[r * 2 + 0] = s0;
  out[r * 2 + 1] = s1;
  out[514 + r]   = sv;                           // value
  if (r < 2) out[512 + r] = __expf(log_std[r]);  // std
}

// ---------------------------------------------------------------------------
// Launcher. Output layout (floats): actor_mean[512] | std[2] | value[256] |
// hT[32768] | cT[32768]  => 66306 total.
// ---------------------------------------------------------------------------
extern "C" void kernel_launch(void* const* d_in, const int* in_sizes, int n_in,
                              void* d_out, int out_size, void* d_ws, size_t ws_size,
                              hipStream_t stream) {
  const float* x       = (const float*)d_in[0];
  const float* h0      = (const float*)d_in[1];
  const float* c0      = (const float*)d_in[2];
  const float* Wx      = (const float*)d_in[3];
  const float* Wh      = (const float*)d_in[4];
  const float* b       = (const float*)d_in[5];
  const float* Wa1     = (const float*)d_in[6];
  const float* ba1     = (const float*)d_in[7];
  const float* Wa2     = (const float*)d_in[8];
  const float* ba2     = (const float*)d_in[9];
  const float* log_std = (const float*)d_in[10];
  const float* Wc1     = (const float*)d_in[11];
  const float* bc1     = (const float*)d_in[12];
  const float* Wc2     = (const float*)d_in[13];
  const float* bc2     = (const float*)d_in[14];

  char* ws = (char*)d_ws;
  half_t* wx_h   = (half_t*)(ws);               // 720*512*2   = 737280 B
  half_t* wh_h   = (half_t*)(ws + 737280);      // 128*512*2   = 131072 B
  half_t* wa1_h  = (half_t*)(ws + 868352);      // 128*128*2   =  32768 B
  half_t* wc1_h  = (half_t*)(ws + 901120);      // 128*128*2   =  32768 B
  half_t* feat_h = (half_t*)(ws + 933888);      // 256*128*2   =  65536 B
  float*  z1     = (float*)(ws + 999424);       // 256*128*4   = 131072 B
  float*  zc1    = (float*)(ws + 1130496);      // 256*128*4   = 131072 B
  float*  xW     = (float*)(ws + 1261568);      // 131072*512*4 = 256 MB

  float* out = (float*)d_out;
  float* out_hT = out + 770;
  float* out_cT = out + 33538;

  // 1) weight conversion
  k_cvt<<<1824, 256, 0, stream>>>(Wx, Wh, Wa1, Wc1, wx_h, wh_h, wa1_h, wc1_h);
  // 2) input projection GEMM
  k_gemm<<<dim3(4, 1024), 256, 0, stream>>>(x, wx_h, b, xW);
  // 3) recurrence (dynamic LDS: Whsw 131072 + hh 4352 + hf/cf 16384 = 151808 B)
  k_lstm<<<16, 256, 151808, stream>>>(xW, wh_h, h0, c0, out_hT, out_cT, feat_h);
  // 4) hidden MLPs
  k_mlp1<<<2, 256, 0, stream>>>(feat_h, wa1_h, wc1_h, ba1, bc1, z1, zc1);
  // 5) final heads
  k_head2<<<1, 256, 0, stream>>>(z1, zc1, Wa2, ba2, log_std, Wc2, bc2, out);
}
